// ConditionalSelfAttention_58050777973123
// MI455X (gfx1250) — matrix operand order repaired and checked by reading, not compiled, not run on hardware
//
#include <hip/hip_runtime.h>
#include <math.h>

typedef __attribute__((ext_vector_type(16))) _Float16 v16h;
typedef __attribute__((ext_vector_type(8)))  float    v8f;

#define NB   8
#define CCH  256
#define NPIX 4096
#define QKD  32
#define LCH  32
#define LOG2E 1.4426950408889634f

union HFrag { v16h v; uint4 q[2]; };

// ---------------------------------------------------------------------------
// 16-lane XOR-butterfly reductions. Prefer v_permlane16_b32 (pure VALU) over
// __shfl_xor (which lowers to ds_bpermute_b32 and stalls on DScnt).
// sel nibbles encode lane^k within each row of 16 lanes.
// ---------------------------------------------------------------------------
#if __has_builtin(__builtin_amdgcn_permlane16)
__device__ __forceinline__ float lanexor(float v, unsigned s0, unsigned s1) {
    int x = __float_as_int(v);
    int y = __builtin_amdgcn_permlane16(x, x, s0, s1, false, false);
    return __int_as_float(y);
}
__device__ __forceinline__ float redmax16(float v) {
    v = fmaxf(v, lanexor(v, 0x67452301u, 0xEFCDAB89u));  // ^1
    v = fmaxf(v, lanexor(v, 0x54761032u, 0xDCFE98BAu));  // ^2
    v = fmaxf(v, lanexor(v, 0x32107654u, 0xBA98FEDCu));  // ^4
    v = fmaxf(v, lanexor(v, 0xFEDCBA98u, 0x76543210u));  // ^8
    return v;
}
__device__ __forceinline__ float redsum16(float v) {
    v += lanexor(v, 0x67452301u, 0xEFCDAB89u);
    v += lanexor(v, 0x54761032u, 0xDCFE98BAu);
    v += lanexor(v, 0x32107654u, 0xBA98FEDCu);
    v += lanexor(v, 0xFEDCBA98u, 0x76543210u);
    return v;
}
#else
__device__ __forceinline__ float redmax16(float v) {
    v = fmaxf(v, __shfl_xor(v, 1));
    v = fmaxf(v, __shfl_xor(v, 2));
    v = fmaxf(v, __shfl_xor(v, 4));
    v = fmaxf(v, __shfl_xor(v, 8));
    return v;
}
__device__ __forceinline__ float redsum16(float v) {
    v += __shfl_xor(v, 1);
    v += __shfl_xor(v, 2);
    v += __shfl_xor(v, 4);
    v += __shfl_xor(v, 8);
    return v;
}
#endif

// ---------------------------------------------------------------------------
// Kernel 1: QKV projection (1x1 conv == per-pixel linear), fp32 -> fp16.
// Q,K stored (B, N, 32) row-major; V stored (B, C, N).
// ---------------------------------------------------------------------------
__global__ __launch_bounds__(256)
void proj_qkv_kernel(const float* __restrict__ x,
                     const float* __restrict__ Wq, const float* __restrict__ bq,
                     const float* __restrict__ Wk, const float* __restrict__ bk,
                     const float* __restrict__ Wv, const float* __restrict__ bv,
                     _Float16* __restrict__ qh, _Float16* __restrict__ kh,
                     _Float16* __restrict__ vh)
{
    __shared__ float xs[CCH * 32];           // 32 KB: x[:, i0:i0+32]
    const int b  = blockIdx.y;
    const int i0 = blockIdx.x * 32;
    const int tid = threadIdx.x;

    for (int idx = tid; idx < CCH * 32; idx += 256) {
        int c = idx >> 5, i = idx & 31;
        xs[idx] = x[((size_t)(b * CCH + c)) * NPIX + i0 + i];
    }
    __syncthreads();

    const int og = tid >> 5;                 // uniform per wave
    const int i  = tid & 31;
    const int ii = i0 + i;
    for (int row8 = 0; row8 < 320; row8 += 8) {
        const int o = row8 + og;
        const float* W;
        float bias;
        int kind;
        if (o < 32)      { W = Wq + (size_t)o * CCH;        bias = bq[o];      kind = 0; }
        else if (o < 64) { W = Wk + (size_t)(o - 32) * CCH; bias = bk[o - 32]; kind = 1; }
        else             { W = Wv + (size_t)(o - 64) * CCH; bias = bv[o - 64]; kind = 2; }
        float acc = bias;
        #pragma unroll 8
        for (int c = 0; c < CCH; ++c)
            acc += W[c] * xs[c * 32 + i];
        if (kind == 0)
            qh[((size_t)(b * NPIX + ii)) * QKD + o] = (_Float16)acc;
        else if (kind == 1)
            kh[((size_t)(b * NPIX + ii)) * QKD + (o - 32)] = (_Float16)acc;
        else
            vh[((size_t)(b * CCH + (o - 64))) * NPIX + ii] = (_Float16)acc;
    }
}

// ---------------------------------------------------------------------------
// Kernel 2: flash-style attention. One WG = 64 query rows of one batch.
// Key tile KT=64. 8 waves: wave w -> S tiles (mtile = w&3, npair = w>>2 gives
// key sub-columns npair*32..+31), O chunk cols w*32..+31.
// Softmax runs in log2 domain (S pre-scaled by log2(e), exp2f throughout).
// ---------------------------------------------------------------------------
__global__ __launch_bounds__(256)
void flash_attn_kernel(const float* __restrict__ x,
                       const _Float16* __restrict__ qh,
                       const _Float16* __restrict__ kh,
                       const _Float16* __restrict__ vh,
                       const float* __restrict__ gamma,
                       float* __restrict__ out)
{
    __shared__ __align__(16) _Float16 Pbuf[64 * 64];   // 8 KB P tile (f16)
    __shared__ __align__(16) float pmax[2][64];
    __shared__ __align__(16) float psum[2][64];
    __shared__ __align__(16) float sscale[64];
    __shared__ __align__(16) float stm[64];
    __shared__ __align__(16) float stl[64];

    const int b    = blockIdx.y;
    const int i0   = blockIdx.x * 64;
    const int tid  = threadIdx.x;
    const int wave = tid >> 5;
    const int lane = tid & 31;
    const int mtile = wave & 3;
    const int npair = wave >> 2;
    const int half  = lane >> 4;
    const int l15   = lane & 15;
    const int cbase = wave * 32;

    if (tid < 64) { stm[tid] = -INFINITY; stl[tid] = 0.0f; }

    // A-fragment of Q (16x32 f16): lane holds row i, K half*16..half*16+15
    HFrag aq;
    {
        const _Float16* qb = qh + ((size_t)(b * NPIX + i0 + mtile * 16 + l15)) * QKD;
        aq.q[0] = *(const uint4*)(qb + half * 16);
        aq.q[1] = *(const uint4*)(qb + half * 16 + 8);
    }

    const v8f zero = {0.f, 0.f, 0.f, 0.f, 0.f, 0.f, 0.f, 0.f};
    v8f O[4][2];
    #pragma unroll
    for (int mt = 0; mt < 4; ++mt)
        #pragma unroll
        for (int nt = 0; nt < 2; ++nt)
            O[mt][nt] = zero;

    __syncthreads();

    for (int j0 = 0; j0 < NPIX; j0 += 64) {
        // ---- (1) two S = Q x K^T 16x16 tiles per wave ----
        HFrag bk0, bk1;
        {
            const _Float16* kb = kh + ((size_t)(b * NPIX + j0 + npair * 32 + l15)) * QKD
                                    + half * 16;
            bk0.q[0] = *(const uint4*)(kb);
            bk0.q[1] = *(const uint4*)(kb + 8);
            const _Float16* kb1 = kb + 16 * QKD;
            bk1.q[0] = *(const uint4*)(kb1);
            bk1.q[1] = *(const uint4*)(kb1 + 8);
        }
        v8f S0 = __builtin_amdgcn_wmma_f32_16x16x32_f16(
            false, aq.v, false, bk0.v, (short)0, zero, false, false);
        v8f S1 = __builtin_amdgcn_wmma_f32_16x16x32_f16(
            false, aq.v, false, bk1.v, (short)0, zero, false, false);

        float rmax[8];
        #pragma unroll
        for (int r = 0; r < 8; ++r) {
            S0[r] = S0[r] * LOG2E;           // move to log2 domain
            S1[r] = S1[r] * LOG2E;
            rmax[r] = redmax16(fmaxf(S0[r], S1[r]));
        }
        if (l15 == 0) {
            #pragma unroll
            for (int r = 0; r < 8; ++r)
                pmax[npair][mtile * 16 + half * 8 + r] = rmax[r];
        }
        __syncthreads();  // barrier A

        // ---- (2) online softmax partials ----
        float mnew[8], resc[8];
        #pragma unroll
        for (int r = 0; r < 8; ++r) {
            const int row = mtile * 16 + half * 8 + r;
            const float mp = stm[row];
            const float mn = fmaxf(mp, fmaxf(pmax[0][row], pmax[1][row]));
            mnew[r] = mn;
            resc[r] = exp2f(mp - mn);
            const float p0 = exp2f(S0[r] - mn);
            const float p1 = exp2f(S1[r] - mn);
            Pbuf[row * 64 + npair * 32 + l15]      = (_Float16)p0;
            Pbuf[row * 64 + npair * 32 + 16 + l15] = (_Float16)p1;
            const float s = redsum16(p0 + p1);
            if (l15 == 0) {
                psum[npair][row] = s;
                if (npair == 0) sscale[row] = resc[r];
            }
        }
        __syncthreads();  // barrier B

        // ---- (3) state update, O rescale, O += P x V ----
        if (npair == 0 && l15 == 0) {
            #pragma unroll
            for (int r = 0; r < 8; ++r) {
                const int row = mtile * 16 + half * 8 + r;
                stl[row] = stl[row] * resc[r] + psum[0][row] + psum[1][row];
                stm[row] = mnew[r];
            }
        }

        // V B-fragments for both k-chunks and both C sub-tiles (held in regs)
        HFrag bV[2][2];
        #pragma unroll
        for (int kc = 0; kc < 2; ++kc) {
            #pragma unroll
            for (int nt = 0; nt < 2; ++nt) {
                const _Float16* vb = vh
                    + ((size_t)(b * CCH + cbase + nt * 16 + l15)) * NPIX
                    + j0 + kc * 32 + half * 16;
                bV[kc][nt].q[0] = *(const uint4*)(vb);
                bV[kc][nt].q[1] = *(const uint4*)(vb + 8);
            }
        }

        #pragma unroll
        for (int mt = 0; mt < 4; ++mt) {
            #pragma unroll
            for (int r = 0; r < 8; ++r) {
                const float sc = sscale[mt * 16 + half * 8 + r];
                O[mt][0][r] = O[mt][0][r] * sc;
                O[mt][1][r] = O[mt][1][r] * sc;
            }
        }

        #pragma unroll
        for (int mt = 0; mt < 4; ++mt) {
            #pragma unroll
            for (int kc = 0; kc < 2; ++kc) {
                HFrag aP;
                const _Float16* pb = Pbuf + (mt * 16 + l15) * 64 + kc * 32;
                aP.q[0] = *(const uint4*)(pb + half * 16);
                aP.q[1] = *(const uint4*)(pb + half * 16 + 8);
                #pragma unroll
                for (int nt = 0; nt < 2; ++nt) {
                    O[mt][nt] = __builtin_amdgcn_wmma_f32_16x16x32_f16(
                        false, aP.v, false, bV[kc][nt].v, (short)0, O[mt][nt],
                        false, false);
                }
            }
        }
    }
    __syncthreads();

    // ---- epilogue: out = gamma * (O / l) + x ----
    const float g = gamma[0];
    #pragma unroll
    for (int mt = 0; mt < 4; ++mt) {
        float linv[8];
        #pragma unroll
        for (int r = 0; r < 8; ++r)
            linv[r] = 1.0f / stl[mt * 16 + half * 8 + r];
        #pragma unroll
        for (int nt = 0; nt < 2; ++nt) {
            const int c = cbase + nt * 16 + l15;
            const size_t rowoff = (size_t)i0 + mt * 16 + half * 8;
            const size_t xoff = ((size_t)(b * CCH + c)) * NPIX + rowoff;
            const size_t ooff = ((size_t)(b * (CCH + LCH)) + c) * NPIX + rowoff;
            float4 x0 = *(const float4*)(x + xoff);
            float4 x1 = *(const float4*)(x + xoff + 4);
            float4 o0, o1;
            o0.x = g * (O[mt][nt][0] * linv[0]) + x0.x;
            o0.y = g * (O[mt][nt][1] * linv[1]) + x0.y;
            o0.z = g * (O[mt][nt][2] * linv[2]) + x0.z;
            o0.w = g * (O[mt][nt][3] * linv[3]) + x0.w;
            o1.x = g * (O[mt][nt][4] * linv[4]) + x1.x;
            o1.y = g * (O[mt][nt][5] * linv[5]) + x1.y;
            o1.z = g * (O[mt][nt][6] * linv[6]) + x1.z;
            o1.w = g * (O[mt][nt][7] * linv[7]) + x1.w;
            *(float4*)(out + ooff)     = o0;
            *(float4*)(out + ooff + 4) = o1;
        }
    }
}

// ---------------------------------------------------------------------------
// Kernel 3: label branch: softmax(label) @ We^T + be, broadcast to (H,W),
// written to output channels 256..287.
// ---------------------------------------------------------------------------
__global__ __launch_bounds__(256)
void label_branch_kernel(const float* __restrict__ label,
                         const float* __restrict__ We,
                         const float* __restrict__ be,
                         float* __restrict__ out)
{
    __shared__ float sl[32];
    __shared__ float su[32];
    const int b   = blockIdx.x;
    const int tid = threadIdx.x;

    if (tid < 32) {
        const float v = label[b * 32 + tid];
        float m = v;
        for (int off = 1; off < 32; off <<= 1) m = fmaxf(m, __shfl_xor(m, off));
        const float e = __expf(v - m);
        float s = e;
        for (int off = 1; off < 32; off <<= 1) s += __shfl_xor(s, off);
        sl[tid] = e / s;
    }
    __syncthreads();
    if (tid < 32) {
        float acc = be[tid];
        #pragma unroll
        for (int c = 0; c < 32; ++c) acc += We[tid * 32 + c] * sl[c];
        su[tid] = acc;
    }
    __syncthreads();
    for (int o = 0; o < 32; ++o) {
        const float u = su[o];
        for (int idx = tid; idx < NPIX; idx += 256)
            out[((size_t)(b * (CCH + LCH)) + CCH + o) * NPIX + idx] = u;
    }
}

// ---------------------------------------------------------------------------
extern "C" void kernel_launch(void* const* d_in, const int* in_sizes, int n_in,
                              void* d_out, int out_size, void* d_ws, size_t ws_size,
                              hipStream_t stream) {
    (void)in_sizes; (void)n_in; (void)out_size; (void)ws_size;

    const float* x     = (const float*)d_in[0];
    const float* label = (const float*)d_in[1];
    const float* Wq    = (const float*)d_in[2];
    const float* bq    = (const float*)d_in[3];
    const float* Wk    = (const float*)d_in[4];
    const float* bk    = (const float*)d_in[5];
    const float* Wv    = (const float*)d_in[6];
    const float* bv    = (const float*)d_in[7];
    const float* gamma = (const float*)d_in[8];
    const float* We    = (const float*)d_in[9];
    const float* be    = (const float*)d_in[10];
    float* out = (float*)d_out;

    _Float16* qh = (_Float16*)d_ws;                       //  2 MB
    _Float16* kh = qh + (size_t)NB * NPIX * QKD;          //  2 MB
    _Float16* vh = kh + (size_t)NB * NPIX * QKD;          // 16 MB

    proj_qkv_kernel<<<dim3(NPIX / 32, NB), 256, 0, stream>>>(
        x, Wq, bq, Wk, bk, Wv, bv, qh, kh, vh);
    flash_attn_kernel<<<dim3(NPIX / 64, NB), 256, 0, stream>>>(
        x, qh, kh, vh, gamma, out);
    label_branch_kernel<<<NB, 256, 0, stream>>>(label, We, be, out);
}
